// SocialGCN_GBSR_84602265796764
// MI455X (gfx1250) — matrix operand order repaired and checked by reading, not compiled, not run on hardware
//
#include <hip/hip_runtime.h>
#include <hip/hip_bf16.h>
#include <math.h>

typedef __attribute__((ext_vector_type(16))) _Float16 v16h;
typedef __attribute__((ext_vector_type(8)))  float    v8f;

#define LDK 66   // padded LDS pitch for transposed weights (conflict-free)

// ---------------------------------------------------------------------------
// Y[nrows,64] = X[nrows,64] @ W[64,64] (+bias) using v_wmma_f32_16x16x32_f16.
// One wave computes a 16x64 tile: 4 accumulators x 2 K-steps = 8 WMMAs.
// f32 inputs converted to f16 in-register while building fragments.
// ---------------------------------------------------------------------------
__global__ void gemm64_wmma(const float* __restrict__ X,
                            const float* __restrict__ W,
                            const float* __restrict__ bias,
                            float* __restrict__ Y, int nrows) {
    __shared__ _Float16 Wt[64 * LDK];          // Wt[n*LDK + k] = (f16)W[k*64+n]
    const int tid = threadIdx.x;

    #pragma unroll
    for (int t = 0; t < 16; ++t) {             // 256 thr * 16 = 4096 elems
        int g = tid + t * 256;
        int k = g >> 6, n = g & 63;
        Wt[n * LDK + k] = (_Float16)W[g];
    }
    __syncthreads();

    const int wave = tid >> 5;
    const int lane = tid & 31;
    const int m0   = (blockIdx.x * 8 + wave) * 16;   // wave-uniform
    if (m0 >= nrows) return;                          // uniform exit, EXEC full
    const bool fullTile = (m0 + 16 <= nrows);         // wave-uniform

    // ---- A fragments (16x32 f16 layout), two K-blocks -----------------
    const int mrow  = min(m0 + (lane & 15), nrows - 1);
    const int half8 = (lane >> 4) << 4;               // 0 or 16
    const float* xb = X + (size_t)mrow * 64 + half8;  // imm-offset base
    __builtin_prefetch(X + (size_t)min(m0 + 128, nrows - 1) * 64, 0, 0);

    v16h a[2];
    #pragma unroll
    for (int kb = 0; kb < 2; ++kb) {
        #pragma unroll
        for (int j = 0; j < 4; ++j) {
            float2 p = *(const float2*)(xb + kb * 32 + 2 * j);
            a[kb][2*j]     = (_Float16)p.x;
            a[kb][2*j + 1] = (_Float16)p.y;
            float2 q = *(const float2*)(xb + kb * 32 + 2 * j + 8);
            a[kb][8 + 2*j]     = (_Float16)q.x;
            a[kb][8 + 2*j + 1] = (_Float16)q.y;
        }
    }

    // ---- B fragments from LDS + WMMA ----------------------------------
    const int nlane = lane & 15;
    const int koff  = (lane >> 4) << 4;               // 0 or 16
    const int mbase = m0 + ((lane >> 4) << 3);

    #pragma unroll
    for (int nb = 0; nb < 4; ++nb) {
        const int n0 = nb * 16;
        v8f c = {0.f, 0.f, 0.f, 0.f, 0.f, 0.f, 0.f, 0.f};
        #pragma unroll
        for (int kb = 0; kb < 2; ++kb) {
            const _Float16* colp = &Wt[(n0 + nlane) * LDK + kb * 32 + koff];
            v16h bf;
            #pragma unroll
            for (int j = 0; j < 8; ++j) {
                bf[2*j]     = colp[2*j];
                bf[2*j + 1] = colp[2*j + 1];
            }
            c = __builtin_amdgcn_wmma_f32_16x16x32_f16(
                    false, a[kb], false, bf, (short)0, c, false, false);
        }
        const int nc = n0 + nlane;
        const float bb = bias ? bias[nc] : 0.0f;
        float* yp = Y + (size_t)mbase * 64 + nc;      // stores at imm offsets
        if (fullTile) {
            #pragma unroll
            for (int r = 0; r < 8; ++r) yp[r * 64] = c[r] + bb;
        } else {
            #pragma unroll
            for (int r = 0; r < 8; ++r)
                if (mbase + r < nrows) yp[r * 64] = c[r] + bb;
        }
    }
}

// ---------------------------------------------------------------------------
// mask[e] = sigmoid( relu(A[row]+B[col]+l1_b) . l2_W + l2_b ).  Wave per edge;
// lane handles channels {lane, lane+32}; wave32 shfl_xor reduction.
// ---------------------------------------------------------------------------
__global__ void edge_mask_kernel(const float* __restrict__ A,
                                 const float* __restrict__ B,
                                 const float* __restrict__ l1b,
                                 const float* __restrict__ l2W,
                                 const float* __restrict__ l2b,
                                 const int* __restrict__ row,
                                 const int* __restrict__ col,
                                 float* __restrict__ mask, int E) {
    const int lane = threadIdx.x & 31;
    const int e    = blockIdx.x * 8 + (threadIdx.x >> 5);
    if (e >= E) return;                               // wave-uniform
    const size_t r = (size_t)row[e] * 64;
    const size_t c = (size_t)col[e] * 64;

    float v0 = A[r + lane]      + B[c + lane]      + l1b[lane];
    float v1 = A[r + lane + 32] + B[c + lane + 32] + l1b[lane + 32];
    v0 = fmaxf(v0, 0.0f) * l2W[lane];
    v1 = fmaxf(v1, 0.0f) * l2W[lane + 32];
    float s = v0 + v1;
    #pragma unroll
    for (int off = 16; off > 0; off >>= 1) s += __shfl_xor(s, off, 32);
    if (lane == 0) mask[e] = 1.0f / (1.0f + __expf(-(s + l2b[0])));
}

// h[col] += mask[e] * P[row]  — wave per edge, f32 atomics into L2.
__global__ void scatter_kernel(const float* __restrict__ P,
                               const float* __restrict__ mask,
                               const int* __restrict__ row,
                               const int* __restrict__ col,
                               float* __restrict__ h, int E) {
    const int lane = threadIdx.x & 31;
    const int e    = blockIdx.x * 8 + (threadIdx.x >> 5);
    if (e >= E) return;
    const size_t r = (size_t)row[e] * 64;
    const size_t c = (size_t)col[e] * 64;
    const float mk = mask[e];
    atomicAdd(&h[c + lane],      mk * P[r + lane]);
    atomicAdd(&h[c + lane + 32], mk * P[r + lane + 32]);
}

// h[i..i+3] = bias (64-channel broadcast), float4 vectorized (n4 = NH/4)
__global__ void init_bias_kernel(float4* __restrict__ h,
                                 const float4* __restrict__ b4, int n4) {
    int i = blockIdx.x * blockDim.x + threadIdx.x;
    if (i < n4) h[i] = b4[i & 15];
}

__global__ void relu_kernel(float4* __restrict__ h, int n4) {
    int i = blockIdx.x * blockDim.x + threadIdx.x;
    if (i < n4) {
        float4 v = h[i];
        v.x = fmaxf(v.x, 0.f); v.y = fmaxf(v.y, 0.f);
        v.z = fmaxf(v.z, 0.f); v.w = fmaxf(v.w, 0.f);
        h[i] = v;
    }
}

// out = (first ? 0 : out) + 0.25*src   (final mean over {emb0,h1,h2,h3})
__global__ void accum_kernel(float4* __restrict__ out,
                             const float4* __restrict__ src, int n4, int first) {
    int i = blockIdx.x * blockDim.x + threadIdx.x;
    if (i < n4) {
        float4 s = src[i];
        float4 o = first ? make_float4(0.f, 0.f, 0.f, 0.f) : out[i];
        o.x += 0.25f * s.x; o.y += 0.25f * s.y;
        o.z += 0.25f * s.z; o.w += 0.25f * s.w;
        out[i] = o;
    }
}

// ---------------------------------------------------------------------------
extern "C" void kernel_launch(void* const* d_in, const int* in_sizes, int n_in,
                              void* d_out, int out_size, void* d_ws, size_t ws_size,
                              hipStream_t stream) {
    const float* x    = (const float*)d_in[0];
    const int*   ei   = (const int*)  d_in[1];
    const float* W_in = (const float*)d_in[2];
    const float* b_in = (const float*)d_in[3];
    const float* l1W  = (const float*)d_in[4];
    const float* l1b  = (const float*)d_in[5];
    const float* l2W  = (const float*)d_in[6];
    const float* l2b  = (const float*)d_in[7];
    const float* gW   = (const float*)d_in[8];
    const float* gb   = (const float*)d_in[9];

    const int H  = 64;
    const int N  = in_sizes[0] / H;
    const int E  = in_sizes[1] / 2;
    const size_t NH = (size_t)N * H;
    const int NH4 = (int)(NH / 4);

    float* ws   = (float*)d_ws;
    float* emb0 = ws;
    float* buf0 = ws + 1 * NH;   // A matrix, then P (msg projection)
    float* buf1 = ws + 2 * NH;   // B matrix, then h ping
    float* buf2 = ws + 3 * NH;   // h pong
    float* mask = ws + 4 * NH;
    float* out  = (float*)d_out;

    const int* row = ei;
    const int* col = ei + E;

    dim3 tb(256);
    const int tiles      = (N + 15) / 16;
    const int gemmBlocks = (tiles + 7) / 8;
    const int vecBlocks  = (NH4 + 255) / 256;
    const int edgeBlocks = (E + 7) / 8;

    // input projection + graph-learner projections
    gemm64_wmma<<<gemmBlocks, tb, 0, stream>>>(x,    W_in,        b_in,    emb0, N);
    gemm64_wmma<<<gemmBlocks, tb, 0, stream>>>(emb0, l1W,         nullptr, buf0, N);
    gemm64_wmma<<<gemmBlocks, tb, 0, stream>>>(emb0, l1W + 64*64, nullptr, buf1, N);
    edge_mask_kernel<<<edgeBlocks, tb, 0, stream>>>(buf0, buf1, l1b, l2W, l2b,
                                                    row, col, mask, E);
    accum_kernel<<<vecBlocks, tb, 0, stream>>>((float4*)out, (const float4*)emb0,
                                               NH4, 1);

    // GCN layers
    const float* hcur = emb0;
    float* hnext_tab[3] = {buf1, buf2, buf1};
    for (int li = 0; li < 3; ++li) {
        float* hnext = hnext_tab[li];
        gemm64_wmma<<<gemmBlocks, tb, 0, stream>>>(hcur, gW + li*64*64, nullptr,
                                                   buf0, N);
        init_bias_kernel<<<vecBlocks, tb, 0, stream>>>((float4*)hnext,
                                                       (const float4*)(gb + li*64),
                                                       NH4);
        scatter_kernel<<<edgeBlocks, tb, 0, stream>>>(buf0, mask, row, col,
                                                      hnext, E);
        if (li < 2) relu_kernel<<<vecBlocks, tb, 0, stream>>>((float4*)hnext, NH4);
        accum_kernel<<<vecBlocks, tb, 0, stream>>>((float4*)out,
                                                   (const float4*)hnext, NH4, 0);
        hcur = hnext;
    }
}
